// MQAModel_24481313587322
// MI455X (gfx1250) — matrix operand order repaired and checked by reading, not compiled, not run on hardware
//
#include <hip/hip_runtime.h>
#include <hip/hip_bf16.h>

// ---------------------------------------------------------------------------
// GVP-GNN forward for MI455X (gfx1250, wave32, WMMA).
// ws matmuls on v_wmma_f32_16x16x32_f16; M = 32 edges per wave in the message
// kernel (two 16-row A tiles sharing every B fragment -> half the weight
// traffic, 2 independent WMMAs per load). B fragments are software-pipelined
// (depth-1 prefetch) so WMMAs don't sit behind s_wait_loadcnt 0.
// Params assumed flattened in jax pytree (sorted-dict-key) order into d_in[0].
// ---------------------------------------------------------------------------

typedef _Float16 half4  __attribute__((ext_vector_type(4)));
typedef _Float16 half8  __attribute__((ext_vector_type(8)));
typedef _Float16 half16 __attribute__((ext_vector_type(16)));
typedef float    float4v __attribute__((ext_vector_type(4)));
typedef float    float8 __attribute__((ext_vector_type(8)));

#define NN 20000
#define NE 320000
#define NG 64
#define EPSf 1e-8f

struct MsgW {
  const float *wh0, *wsb0, *wv0;
  const float *wh1, *wsb1, *wv1;
  const float *wh2, *wsb2, *wv2;
  const _Float16 *ws0h, *ws1h, *ws2h;   // Kpad = 288, 128, 128 ; rows padded to 112
};

struct FFW {
  const float *wh0, *wsb0, *wv0;
  const float *wh1, *wsb1, *wv1;
  const _Float16 *ws0h, *ws1h;          // Kpad = 160 (rows 400), 448 (rows 112)
  const float *ln0g, *ln0b, *ln1g, *ln1b;
};

// ---------------------------------------------------------------------------
// Lane mapping per CDNA5 ISA 7.12.2 (16-bit A, f32 C):
//   A frag half h (lane l): k = (h&8?16:0) + (l&16?8:0) + (h&7), row = l&15
//   B frag half h (lane l): k = (h&8?16:0) + (l&16?8:0) + (h&7),  n = n0 + (l&15)
//   C frag elem g (lane l): m = g + (l&16?8:0),   n = n0 + (l&15)
// ---------------------------------------------------------------------------

// Single 16-row tile GEMM with depth-1 B prefetch.
__device__ inline float8 wmma_tile(const _Float16* A, int Kpad,
                                   const _Float16* __restrict__ W,
                                   int n0, int lane) {
  float8 acc = {0.f,0.f,0.f,0.f,0.f,0.f,0.f,0.f};
  const int row = lane & 15;
  const int ak  = (lane & 16) ? 8 : 0;
  const int wk  = (lane & 16) ? 8 : 0;
  const _Float16* ar = A + row * Kpad + ak;
  const _Float16* wr = W + (size_t)(n0 + row) * Kpad + wk;
  union F { half16 v; half8 h[2]; };
  F b;
  b.h[0] = *(const half8*)(wr);
  b.h[1] = *(const half8*)(wr + 16);
  for (int kc = 0; kc < Kpad; kc += 32) {
    F a, bn;
    a.h[0] = *(const half8*)(ar + kc);
    a.h[1] = *(const half8*)(ar + kc + 16);
    if (kc + 32 < Kpad) {
      bn.h[0] = *(const half8*)(wr + kc + 32);
      bn.h[1] = *(const half8*)(wr + kc + 48);
    }
    acc = __builtin_amdgcn_wmma_f32_16x16x32_f16(false, a.v, false, b.v,
                                                 (short)0, acc, false, false);
    if (kc + 32 < Kpad) b = bn;
  }
  return acc;
}

// Dual 16-row tile GEMM: two A tiles share every B fragment (2 WMMAs / load),
// with depth-1 B prefetch.
__device__ inline void wmma_tile2(const _Float16* A0, const _Float16* A1, int Kpad,
                                  const _Float16* __restrict__ W,
                                  int n0, int lane, float8& out0, float8& out1) {
  float8 c0 = {0.f,0.f,0.f,0.f,0.f,0.f,0.f,0.f};
  float8 c1 = {0.f,0.f,0.f,0.f,0.f,0.f,0.f,0.f};
  const int row = lane & 15;
  const int ak  = (lane & 16) ? 8 : 0;
  const int wk  = (lane & 16) ? 8 : 0;
  const _Float16* ar0 = A0 + row * Kpad + ak;
  const _Float16* ar1 = A1 + row * Kpad + ak;
  const _Float16* wr  = W + (size_t)(n0 + row) * Kpad + wk;
  union F { half16 v; half8 h[2]; };
  F b;
  b.h[0] = *(const half8*)(wr);
  b.h[1] = *(const half8*)(wr + 16);
  for (int kc = 0; kc < Kpad; kc += 32) {
    F a0, a1, bn;
    a0.h[0] = *(const half8*)(ar0 + kc);
    a0.h[1] = *(const half8*)(ar0 + kc + 16);
    a1.h[0] = *(const half8*)(ar1 + kc);
    a1.h[1] = *(const half8*)(ar1 + kc + 16);
    if (kc + 32 < Kpad) {
      bn.h[0] = *(const half8*)(wr + kc + 32);
      bn.h[1] = *(const half8*)(wr + kc + 48);
    }
    c0 = __builtin_amdgcn_wmma_f32_16x16x32_f16(false, a0.v, false, b.v,
                                                (short)0, c0, false, false);
    c1 = __builtin_amdgcn_wmma_f32_16x16x32_f16(false, a1.v, false, b.v,
                                                (short)0, c1, false, false);
    if (kc + 32 < Kpad) b = bn;
  }
  out0 = c0; out1 = c1;
}

// ---------------------------------------------------------------------------
// Weight f32 -> f16 with zero padding.  dst [rowsPad][colsPad]
// ---------------------------------------------------------------------------
__global__ void k_cvt16(const float* __restrict__ src, _Float16* __restrict__ dst,
                        int rows, int cols, int rowsPad, int colsPad) {
  int i = blockIdx.x * 256 + threadIdx.x;
  int tot = rowsPad * colsPad;
  if (i >= tot) return;
  int r = i / colsPad, c = i % colsPad;
  float v = (r < rows && c < cols) ? src[(size_t)r * cols + c] : 0.f;
  dst[i] = (_Float16)v;
}

__global__ void k_deg(const int* __restrict__ dst, float* __restrict__ deg) {
  int e = blockIdx.x * 256 + threadIdx.x;
  if (e < NE) atomicAdd(&deg[dst[e]], 1.f);
}
__global__ void k_invdeg(const float* __restrict__ deg, float* __restrict__ idg) {
  int n = blockIdx.x * 256 + threadIdx.x;
  if (n < NN) idg[n] = 1.f / fmaxf(deg[n], 1.f);
}

// ---------------------------------------------------------------------------
// Node embedding: concat(h_V_s, W_s[seq]) -> LN_sv -> GVP(26,3 -> 100,16)
// ---------------------------------------------------------------------------
__global__ void k_node_embed(const float* __restrict__ hVs, const float* __restrict__ hVv,
                             const int* __restrict__ seq,
                             const float* __restrict__ Ws,
                             const float* __restrict__ lng, const float* __restrict__ lnb,
                             const float* __restrict__ wh, const float* __restrict__ ws,
                             const float* __restrict__ wsb, const float* __restrict__ wv,
                             float* __restrict__ S, float* __restrict__ V) {
  int n = blockIdx.x * 256 + threadIdx.x;
  if (n >= NN) return;
  float sin[26];
  for (int i = 0; i < 6; ++i) sin[i] = hVs[(size_t)n * 6 + i];
  int sq = seq[n];
  for (int i = 0; i < 20; ++i) sin[6 + i] = Ws[sq * 20 + i];
  float mu = 0.f; for (int i = 0; i < 26; ++i) mu += sin[i]; mu /= 26.f;
  float var = 0.f; for (int i = 0; i < 26; ++i) { float d = sin[i] - mu; var += d * d; }
  float rstd = rsqrtf(var / 26.f + 1e-5f);
  for (int i = 0; i < 26; ++i) sin[i] = (sin[i] - mu) * rstd * lng[i] + lnb[i];
  float vv[9];
  for (int i = 0; i < 9; ++i) vv[i] = hVv[(size_t)n * 9 + i];
  float acc = 0.f;
  for (int i = 0; i < 3; ++i) {
    float n2 = vv[i*3]*vv[i*3] + vv[i*3+1]*vv[i*3+1] + vv[i*3+2]*vv[i*3+2];
    acc += fmaxf(n2, EPSf);
  }
  float vr = rsqrtf(acc / 3.f);
  for (int i = 0; i < 9; ++i) vv[i] *= vr;
  float vh[48];
  for (int h = 0; h < 16; ++h)
    for (int c = 0; c < 3; ++c) {
      float a = 0.f;
      for (int i = 0; i < 3; ++i) a += wh[h*3+i] * vv[i*3+c];
      vh[h*3+c] = a;
    }
  float scat[42];
  for (int i = 0; i < 26; ++i) scat[i] = sin[i];
  for (int h = 0; h < 16; ++h) {
    float* q = vh + h*3;
    scat[26+h] = sqrtf(fmaxf(q[0]*q[0]+q[1]*q[1]+q[2]*q[2], EPSf));
  }
  for (int o = 0; o < 100; ++o) {
    float a = wsb[o];
    for (int k = 0; k < 42; ++k) a += ws[o*42+k] * scat[k];
    S[(size_t)n*100 + o] = a;
  }
  for (int o = 0; o < 16; ++o)
    for (int c = 0; c < 3; ++c) {
      float a = 0.f;
      for (int h = 0; h < 16; ++h) a += wv[o*16+h] * vh[h*3+c];
      V[(size_t)n*48 + o*3 + c] = a;
    }
}

// ---------------------------------------------------------------------------
// Edge embedding: LN_sv -> GVP(32,1 -> 32,1)
// ---------------------------------------------------------------------------
__global__ void k_edge_embed(const float* __restrict__ hEs, const float* __restrict__ hEv,
                             const float* __restrict__ lng, const float* __restrict__ lnb,
                             const float* __restrict__ wh, const float* __restrict__ ws,
                             const float* __restrict__ wsb, const float* __restrict__ wv,
                             float* __restrict__ ES, float* __restrict__ EV) {
  int e = blockIdx.x * 256 + threadIdx.x;
  if (e >= NE) return;
  __align__(16) float srow[32];
  for (int i = 0; i < 8; ++i)
    *(float4v*)(srow + i*4) = *(const float4v*)(hEs + (size_t)e*32 + i*4);
  float mu = 0.f; for (int i = 0; i < 32; ++i) mu += srow[i]; mu /= 32.f;
  float var = 0.f; for (int i = 0; i < 32; ++i) { float d = srow[i]-mu; var += d*d; }
  float rstd = rsqrtf(var / 32.f + 1e-5f);
  for (int i = 0; i < 32; ++i) srow[i] = (srow[i]-mu)*rstd*lng[i] + lnb[i];
  float vx = hEv[(size_t)e*3], vy = hEv[(size_t)e*3+1], vz = hEv[(size_t)e*3+2];
  float vr = rsqrtf(fmaxf(vx*vx+vy*vy+vz*vz, EPSf));
  vx *= vr; vy *= vr; vz *= vr;
  float w0 = wh[0];
  float hx = w0*vx, hy = w0*vy, hz = w0*vz;
  float vn = sqrtf(fmaxf(hx*hx+hy*hy+hz*hz, EPSf));
  for (int o = 0; o < 32; ++o) {
    float a = wsb[o];
    for (int k = 0; k < 32; ++k) a += ws[o*33+k] * srow[k];
    a += ws[o*33+32] * vn;
    ES[(size_t)e*32 + o] = a;
  }
  float wv0 = wv[0];
  EV[(size_t)e*3]   = wv0*hx;
  EV[(size_t)e*3+1] = wv0*hy;
  EV[(size_t)e*3+2] = wv0*hz;
}

// ---------------------------------------------------------------------------
// Message kernel: 1 wave/block, wave owns 32 edges (two 16-row WMMA tiles).
// 3-GVP stack: ws matmuls via dual-tile WMMA (B reused 2x), vector channels
// via f32 VALU, scatter via global f32 atomics pre-scaled by 1/deg.
// ---------------------------------------------------------------------------
__global__ __launch_bounds__(32) void k_msg(
    const float* __restrict__ S, const float* __restrict__ V,
    const float* __restrict__ ES, const float* __restrict__ EV,
    const int* __restrict__ src, const int* __restrict__ dst,
    const float* __restrict__ invdeg,
    float* __restrict__ DS, float* __restrict__ DV, MsgW W) {
  __shared__ __align__(16) _Float16 sA[32*288];   // A-tiles (rows 0-15 | 16-31)
  __shared__ __align__(16) _Float16 sB[32*128];   // staging for next GVP input
  __shared__ __align__(16) float sMV[32*99];
  __shared__ __align__(16) float sVH[32*99];
  __shared__ __align__(16) float sVO[32*48];
  __shared__ int   sSrc[32], sDst[32];
  __shared__ float sID[32];

  const int lane = threadIdx.x;
  const int e0 = blockIdx.x * 32;

  {
    int e = e0 + lane;
    int sj = src[e], dj = dst[e];
    sSrc[lane] = sj; sDst[lane] = dj; sID[lane] = invdeg[dj];
  }
  __syncthreads();
  // stage scalar A-tile (float4 gathers): [s_src(100)|e_s(32)|s_dst(100)|vn0|0]
  for (int t = lane; t < 32*72; t += 32) {
    int r = t / 72, c0 = (t % 72) * 4;
    float4v val = {0.f,0.f,0.f,0.f};
    if (c0 < 100)      val = *(const float4v*)(S + (size_t)sSrc[r]*100 + c0);
    else if (c0 < 132) val = *(const float4v*)(ES + (size_t)(e0+r)*32 + (c0-100));
    else if (c0 < 232) val = *(const float4v*)(S + (size_t)sDst[r]*100 + (c0-132));
    *(half4*)(sA + r*288 + c0) = __builtin_convertvector(val, half4);
  }
  // stage vector channels: [v_src(16) | e_v(1) | v_dst(16)] x 3
  for (int t = lane; t < 32*99; t += 32) {
    int r = t/99, rem = t%99, i = rem/3, c = rem%3;
    float val;
    if (i < 16)       val = V[(size_t)sSrc[r]*48 + i*3 + c];
    else if (i == 16) val = EV[(size_t)(e0+r)*3 + c];
    else              val = V[(size_t)sDst[r]*48 + (i-17)*3 + c];
    sMV[t] = val;
  }
  __syncthreads();
  // vh0 = wh0(33x33) @ mv
  for (int t = lane; t < 32*99; t += 32) {
    int r = t/99, rem = t%99, h = rem/3, c = rem%3;
    float acc = 0.f;
    const float* whr = W.wh0 + h*33;
    const float* q = sMV + r*99 + c;
    for (int i = 0; i < 33; ++i) acc += whr[i] * q[i*3];
    sVH[t] = acc;
  }
  __syncthreads();
  // vn0 -> A cols 232..264 ; vout0 (sigmoid-gated) -> VO
  for (int t = lane; t < 32*33; t += 32) {
    int r = t/33, h = t%33;
    const float* q = sVH + r*99 + h*3;
    sA[r*288 + 232 + h] = (_Float16)sqrtf(fmaxf(q[0]*q[0]+q[1]*q[1]+q[2]*q[2], EPSf));
  }
  for (int t = lane; t < 32*16; t += 32) {
    int r = t/16, o = t%16;
    float ax=0.f, ay=0.f, az=0.f;
    const float* wvr = W.wv0 + o*33;
    const float* q = sVH + r*99;
    for (int h = 0; h < 33; ++h) { float g = wvr[h]; ax += g*q[h*3]; ay += g*q[h*3+1]; az += g*q[h*3+2]; }
    float nr = sqrtf(fmaxf(ax*ax+ay*ay+az*az, EPSf));
    float g = 1.f / (1.f + __expf(-nr));
    sVO[r*48+o*3] = ax*g; sVO[r*48+o*3+1] = ay*g; sVO[r*48+o*3+2] = az*g;
  }
  __syncthreads();
  // GEMM0: [32 x 288] x ws0^T -> relu -> sB[32 x 128]
  {
    int rowm = (lane & 16) ? 8 : 0;
    for (int n0 = 0; n0 < 112; n0 += 16) {
      float8 a0, a1;
      wmma_tile2(sA, sA + 16*288, 288, W.ws0h, n0, lane, a0, a1);
      int n = n0 + (lane & 15);
      float bias = (n < 100) ? W.wsb0[n] : 0.f;
#pragma unroll
      for (int g = 0; g < 8; ++g) {
        int m = g + rowm;
        float v0 = (n < 100) ? fmaxf(a0[g] + bias, 0.f) : 0.f;
        float v1 = (n < 100) ? fmaxf(a1[g] + bias, 0.f) : 0.f;
        sB[m*128 + n]      = (_Float16)v0;
        sB[(m+16)*128 + n] = (_Float16)v1;
      }
    }
    for (int t = lane; t < 32*16; t += 32)
      sB[(t/16)*128 + 112 + (t%16)] = (_Float16)0.f;
  }
  __syncthreads();
  // vh1 = wh1(16x16) @ vout0
  for (int t = lane; t < 32*48; t += 32) {
    int r = t/48, rem = t%48, h = rem/3, c = rem%3;
    float acc = 0.f;
    const float* whr = W.wh1 + h*16;
    const float* q = sVO + r*48 + c;
    for (int i = 0; i < 16; ++i) acc += whr[i] * q[i*3];
    sVH[r*48 + rem] = acc;
  }
  __syncthreads();
  // vn1 -> sB cols 100..115 ; vout1 (gated) -> sMV (stride 48)
  for (int t = lane; t < 32*16; t += 32) {
    int r = t/16, h = t%16;
    const float* q = sVH + r*48 + h*3;
    sB[r*128 + 100 + h] = (_Float16)sqrtf(fmaxf(q[0]*q[0]+q[1]*q[1]+q[2]*q[2], EPSf));
  }
  for (int t = lane; t < 32*16; t += 32) {
    int r = t/16, o = t%16;
    float ax=0.f, ay=0.f, az=0.f;
    const float* wvr = W.wv1 + o*16;
    const float* q = sVH + r*48;
    for (int h = 0; h < 16; ++h) { float g = wvr[h]; ax += g*q[h*3]; ay += g*q[h*3+1]; az += g*q[h*3+2]; }
    float nr = sqrtf(fmaxf(ax*ax+ay*ay+az*az, EPSf));
    float g = 1.f / (1.f + __expf(-nr));
    sMV[r*48+o*3] = ax*g; sMV[r*48+o*3+1] = ay*g; sMV[r*48+o*3+2] = az*g;
  }
  __syncthreads();
  // GEMM1: [32 x 128] x ws1^T -> relu -> sA[32 x 128]
  {
    int rowm = (lane & 16) ? 8 : 0;
    for (int n0 = 0; n0 < 112; n0 += 16) {
      float8 a0, a1;
      wmma_tile2(sB, sB + 16*128, 128, W.ws1h, n0, lane, a0, a1);
      int n = n0 + (lane & 15);
      float bias = (n < 100) ? W.wsb1[n] : 0.f;
#pragma unroll
      for (int g = 0; g < 8; ++g) {
        int m = g + rowm;
        float v0 = (n < 100) ? fmaxf(a0[g] + bias, 0.f) : 0.f;
        float v1 = (n < 100) ? fmaxf(a1[g] + bias, 0.f) : 0.f;
        sA[m*128 + n]      = (_Float16)v0;
        sA[(m+16)*128 + n] = (_Float16)v1;
      }
    }
    for (int t = lane; t < 32*16; t += 32)
      sA[(t/16)*128 + 112 + (t%16)] = (_Float16)0.f;
  }
  __syncthreads();
  // vh2 = wh2(16x16) @ vout1
  for (int t = lane; t < 32*48; t += 32) {
    int r = t/48, rem = t%48, h = rem/3, c = rem%3;
    float acc = 0.f;
    const float* whr = W.wh2 + h*16;
    const float* q = sMV + r*48 + c;
    for (int i = 0; i < 16; ++i) acc += whr[i] * q[i*3];
    sVH[r*48 + rem] = acc;
  }
  __syncthreads();
  // vn2 -> sA cols 100..115 ; vout2 (no gate) -> sVO
  for (int t = lane; t < 32*16; t += 32) {
    int r = t/16, h = t%16;
    const float* q = sVH + r*48 + h*3;
    sA[r*128 + 100 + h] = (_Float16)sqrtf(fmaxf(q[0]*q[0]+q[1]*q[1]+q[2]*q[2], EPSf));
  }
  for (int t = lane; t < 32*16; t += 32) {
    int r = t/16, o = t%16;
    float ax=0.f, ay=0.f, az=0.f;
    const float* wvr = W.wv2 + o*16;
    const float* q = sVH + r*48;
    for (int h = 0; h < 16; ++h) { float g = wvr[h]; ax += g*q[h*3]; ay += g*q[h*3+1]; az += g*q[h*3+2]; }
    sVO[r*48+o*3] = ax; sVO[r*48+o*3+1] = ay; sVO[r*48+o*3+2] = az;
  }
  __syncthreads();
  // GEMM2 + mean-scaled scatter to ds / dv
  {
    int rowm = (lane & 16) ? 8 : 0;
    for (int n0 = 0; n0 < 112; n0 += 16) {
      float8 a0, a1;
      wmma_tile2(sA, sA + 16*128, 128, W.ws2h, n0, lane, a0, a1);
      int n = n0 + (lane & 15);
      if (n < 100) {
        float bias = W.wsb2[n];
#pragma unroll
        for (int g = 0; g < 8; ++g) {
          int m = g + rowm;
          atomicAdd(&DS[(size_t)sDst[m]*100 + n],    (a0[g] + bias) * sID[m]);
          atomicAdd(&DS[(size_t)sDst[m+16]*100 + n], (a1[g] + bias) * sID[m+16]);
        }
      }
    }
    for (int t = lane; t < 32*48; t += 32) {
      int r = t/48;
      atomicAdd(&DV[(size_t)sDst[r]*48 + (t%48)], sVO[t] * sID[r]);
    }
  }
}

// ---------------------------------------------------------------------------
// Node update: residual + LN0, feedforward GVPs (WMMA), residual + LN1.
// 1 wave per 16-node tile.
// ---------------------------------------------------------------------------
__global__ __launch_bounds__(32) void k_node_update(
    float* __restrict__ S, float* __restrict__ V,
    const float* __restrict__ DS, const float* __restrict__ DV, FFW W) {
  __shared__ __align__(16) _Float16 uA[16*160];
  __shared__ __align__(16) _Float16 uB[16*448];
  __shared__ __align__(16) float uS[16*100];
  __shared__ __align__(16) float uV[16*48];
  __shared__ __align__(16) float uVH[16*96];
  __shared__ __align__(16) float uVO[16*96];
  __shared__ float uMu[16], uRstd[16], uVr[16];

  const int lane = threadIdx.x;
  const int nd0 = blockIdx.x * 16;

  for (int t = lane; t < 16*25; t += 32) {
    int r = t/25, c0 = (t%25)*4;
    size_t idx = (size_t)(nd0+r)*100 + c0;
    float4v a = *(const float4v*)(S + idx);
    float4v b = *(const float4v*)(DS + idx);
    *(float4v*)(uS + r*100 + c0) = a + b;
  }
  for (int t = lane; t < 16*12; t += 32) {
    int o = t*4;
    float4v a = *(const float4v*)(V + (size_t)nd0*48 + o);
    float4v b = *(const float4v*)(DV + (size_t)nd0*48 + o);
    *(float4v*)(uV + o) = a + b;
  }
  __syncthreads();
  if (lane < 16) {
    const float* row = uS + lane*100;
    float mu = 0.f; for (int c = 0; c < 100; ++c) mu += row[c]; mu *= 0.01f;
    float var = 0.f; for (int c = 0; c < 100; ++c) { float d = row[c]-mu; var += d*d; } var *= 0.01f;
    uMu[lane] = mu; uRstd[lane] = rsqrtf(var + 1e-5f);
  } else {
    int r = lane - 16;
    const float* q = uV + r*48;
    float acc = 0.f;
    for (int i = 0; i < 16; ++i) {
      float n2 = q[i*3]*q[i*3] + q[i*3+1]*q[i*3+1] + q[i*3+2]*q[i*3+2];
      acc += fmaxf(n2, EPSf);
    }
    uVr[r] = rsqrtf(acc / 16.f);
  }
  __syncthreads();
  for (int t = lane; t < 16*100; t += 32) {
    int r = t/100, c = t%100;
    float val = (uS[t] - uMu[r]) * uRstd[r] * W.ln0g[c] + W.ln0b[c];
    uS[t] = val;
    uA[r*160 + c] = (_Float16)val;
  }
  for (int t = lane; t < 16*48; t += 32) { int r = t/48; uV[t] *= uVr[r]; }
  __syncthreads();
  // vh = wh0(32x16) @ v
  for (int t = lane; t < 16*96; t += 32) {
    int r = t/96, rem = t%96, h = rem/3, c = rem%3;
    float acc = 0.f;
    const float* whr = W.wh0 + h*16;
    const float* q = uV + r*48 + c;
    for (int i = 0; i < 16; ++i) acc += whr[i] * q[i*3];
    uVH[t] = acc;
  }
  __syncthreads();
  // vn -> uA cols 100..131, zero 132..159 ; vout0 gated -> uVO
  for (int t = lane; t < 16*32; t += 32) {
    int r = t/32, h = t%32;
    const float* q = uVH + r*96 + h*3;
    uA[r*160 + 100 + h] = (_Float16)sqrtf(fmaxf(q[0]*q[0]+q[1]*q[1]+q[2]*q[2], EPSf));
  }
  for (int t = lane; t < 16*28; t += 32) {
    int r = t/28, c = 132 + t%28;
    uA[r*160 + c] = (_Float16)0.f;
  }
  for (int t = lane; t < 16*32; t += 32) {
    int r = t/32, o = t%32;
    float ax=0.f, ay=0.f, az=0.f;
    const float* wvr = W.wv0 + o*32;
    const float* q = uVH + r*96;
    for (int h = 0; h < 32; ++h) { float g = wvr[h]; ax += g*q[h*3]; ay += g*q[h*3+1]; az += g*q[h*3+2]; }
    float nr = sqrtf(fmaxf(ax*ax+ay*ay+az*az, EPSf));
    float g = 1.f / (1.f + __expf(-nr));
    uVO[r*96+o*3] = ax*g; uVO[r*96+o*3+1] = ay*g; uVO[r*96+o*3+2] = az*g;
  }
  __syncthreads();
  // GEMM ff0: [16x160] x ws0^T(400x160) -> relu -> uB cols 0..399
  {
    int rowm = (lane & 16) ? 8 : 0;
    for (int n0 = 0; n0 < 400; n0 += 16) {
      float8 acc = wmma_tile(uA, 160, W.ws0h, n0, lane);
      int n = n0 + (lane & 15);
      float bias = W.wsb0[n];
#pragma unroll
      for (int g = 0; g < 8; ++g) {
        int m = g + rowm;
        uB[m*448 + n] = (_Float16)fmaxf(acc[g] + bias, 0.f);
      }
    }
  }
  __syncthreads();
  // vh1 = wh1(32x32) @ vout0
  for (int t = lane; t < 16*96; t += 32) {
    int r = t/96, rem = t%96, h = rem/3, c = rem%3;
    float acc = 0.f;
    const float* whr = W.wh1 + h*32;
    const float* q = uVO + r*96 + c;
    for (int i = 0; i < 32; ++i) acc += whr[i] * q[i*3];
    uVH[t] = acc;
  }
  __syncthreads();
  // vn1 -> uB cols 400..431, zero 432..447 ; fv = wv1(16x32)@vh1 added to uV
  for (int t = lane; t < 16*32; t += 32) {
    int r = t/32, h = t%32;
    const float* q = uVH + r*96 + h*3;
    uB[r*448 + 400 + h] = (_Float16)sqrtf(fmaxf(q[0]*q[0]+q[1]*q[1]+q[2]*q[2], EPSf));
  }
  for (int t = lane; t < 16*16; t += 32)
    uB[(t/16)*448 + 432 + (t%16)] = (_Float16)0.f;
  for (int t = lane; t < 16*16; t += 32) {
    int r = t/16, o = t%16;
    float ax=0.f, ay=0.f, az=0.f;
    const float* wvr = W.wv1 + o*32;
    const float* q = uVH + r*96;
    for (int h = 0; h < 32; ++h) { float g = wvr[h]; ax += g*q[h*3]; ay += g*q[h*3+1]; az += g*q[h*3+2]; }
    uV[r*48+o*3] += ax; uV[r*48+o*3+1] += ay; uV[r*48+o*3+2] += az;
  }
  __syncthreads();
  // GEMM ff1: [16x448] x ws1^T -> add into uS (residual)
  {
    int rowm = (lane & 16) ? 8 : 0;
    for (int n0 = 0; n0 < 112; n0 += 16) {
      float8 acc = wmma_tile(uB, 448, W.ws1h, n0, lane);
      int n = n0 + (lane & 15);
      if (n < 100) {
        float bias = W.wsb1[n];
#pragma unroll
        for (int g = 0; g < 8; ++g) {
          int m = g + rowm;
          uS[m*100 + n] += acc[g] + bias;
        }
      }
    }
  }
  __syncthreads();
  if (lane < 16) {
    const float* row = uS + lane*100;
    float mu = 0.f; for (int c = 0; c < 100; ++c) mu += row[c]; mu *= 0.01f;
    float var = 0.f; for (int c = 0; c < 100; ++c) { float d = row[c]-mu; var += d*d; } var *= 0.01f;
    uMu[lane] = mu; uRstd[lane] = rsqrtf(var + 1e-5f);
  } else {
    int r = lane - 16;
    const float* q = uV + r*48;
    float acc = 0.f;
    for (int i = 0; i < 16; ++i) {
      float n2 = q[i*3]*q[i*3] + q[i*3+1]*q[i*3+1] + q[i*3+2]*q[i*3+2];
      acc += fmaxf(n2, EPSf);
    }
    uVr[r] = rsqrtf(acc / 16.f);
  }
  __syncthreads();
  for (int t = lane; t < 16*25; t += 32) {
    int r = t/25, c0 = (t%25)*4;
    float4v x = *(const float4v*)(uS + r*100 + c0);
    float4v o;
#pragma unroll
    for (int j = 0; j < 4; ++j)
      o[j] = (x[j] - uMu[r]) * uRstd[r] * W.ln1g[c0+j] + W.ln1b[c0+j];
    *(float4v*)(S + (size_t)(nd0+r)*100 + c0) = o;
  }
  for (int t = lane; t < 16*12; t += 32) {
    int o = t*4, r = o/48;
    float4v x = *(const float4v*)(uV + o);
    *(float4v*)(V + (size_t)nd0*48 + o) = x * uVr[r];
  }
}

// ---------------------------------------------------------------------------
// Output GVP + scatter-mean pooling (per-node VALU; minor FLOPs)
// ---------------------------------------------------------------------------
__global__ void k_out_pool(const float* __restrict__ S, const float* __restrict__ V,
                           const int* __restrict__ batch,
                           const float* __restrict__ lng, const float* __restrict__ lnb,
                           const float* __restrict__ wh, const float* __restrict__ ws,
                           const float* __restrict__ wsb,
                           float* __restrict__ pooled, float* __restrict__ nb) {
  int n = blockIdx.x * 256 + threadIdx.x;
  if (n >= NN) return;
  __align__(16) float srow[100];
  for (int i = 0; i < 25; ++i)
    *(float4v*)(srow + i*4) = *(const float4v*)(S + (size_t)n*100 + i*4);
  float mu = 0.f; for (int i = 0; i < 100; ++i) mu += srow[i]; mu *= 0.01f;
  float var = 0.f; for (int i = 0; i < 100; ++i) { float d = srow[i]-mu; var += d*d; } var *= 0.01f;
  float rstd = rsqrtf(var + 1e-5f);
  for (int i = 0; i < 100; ++i) srow[i] = (srow[i]-mu)*rstd*lng[i] + lnb[i];
  __align__(16) float vv[48];
  for (int i = 0; i < 12; ++i)
    *(float4v*)(vv + i*4) = *(const float4v*)(V + (size_t)n*48 + i*4);
  float acc = 0.f;
  for (int i = 0; i < 16; ++i) {
    float n2 = vv[i*3]*vv[i*3] + vv[i*3+1]*vv[i*3+1] + vv[i*3+2]*vv[i*3+2];
    acc += fmaxf(n2, EPSf);
  }
  float vr = rsqrtf(acc / 16.f);
  for (int i = 0; i < 48; ++i) vv[i] *= vr;
  float vn[16];
  for (int h = 0; h < 16; ++h) {
    float ax=0.f, ay=0.f, az=0.f;
    for (int i = 0; i < 16; ++i) { float g = wh[h*16+i]; ax += g*vv[i*3]; ay += g*vv[i*3+1]; az += g*vv[i*3+2]; }
    vn[h] = sqrtf(fmaxf(ax*ax+ay*ay+az*az, EPSf));
  }
  int b = batch[n];
  atomicAdd(&nb[b], 1.f);
  for (int o = 0; o < 100; ++o) {
    float a = wsb[o];
    for (int k = 0; k < 100; ++k) a += ws[o*116+k] * srow[k];
    for (int k = 0; k < 16; ++k)  a += ws[o*116+100+k] * vn[k];
    atomicAdd(&pooled[(size_t)b*100 + o], fmaxf(a, 0.f));
  }
}

__global__ void k_dense(const float* __restrict__ pooled, const float* __restrict__ nb,
                        const float* __restrict__ w1, const float* __restrict__ b1,
                        const float* __restrict__ w2, const float* __restrict__ b2,
                        float* __restrict__ out) {
  int g = threadIdx.x;
  if (g >= NG) return;
  float inv = 1.f / fmaxf(nb[g], 1.f);
  float acc = b2[0];
  for (int k = 0; k < 200; ++k) {
    float h = b1[k];
    for (int j = 0; j < 100; ++j) h += w1[k*100+j] * pooled[(size_t)g*100+j] * inv;
    acc += w2[k] * fmaxf(h, 0.f);
  }
  out[g] = acc + 0.5f;
}

// ---------------------------------------------------------------------------
extern "C" void kernel_launch(void* const* d_in, const int* in_sizes, int n_in,
                              void* d_out, int out_size, void* d_ws, size_t ws_size,
                              hipStream_t stream) {
  (void)in_sizes; (void)n_in; (void)out_size; (void)ws_size;
  const float* P   = (const float*)d_in[0];   // flattened params (pytree order)
  const float* hVs = (const float*)d_in[1];
  const float* hVv = (const float*)d_in[2];
  const float* hEs = (const float*)d_in[3];
  const float* hEv = (const float*)d_in[4];
  const int* eidx  = (const int*)d_in[5];
  const int* seq   = (const int*)d_in[6];
  const int* batch = (const int*)d_in[7];
  const int* esrc = eidx;
  const int* edst = eidx + NE;
  float* out = (float*)d_out;

  // --- param offsets (sorted-key pytree order) ---
  size_t off = 0;
  auto take = [&](size_t n) { size_t o = off; off += n; return o; };
  size_t oWS  = take(400);                                       // W_s
  size_t oDb1 = take(200), oDb2 = take(1), oDw1 = take(20000), oDw2 = take(200); // dense
  size_t oGeWh = take(1), oGeWs = take(1056), oGeWsb = take(32), oGeWv = take(1); // gvp_e
  size_t oGoWh = take(256), oGoWs = take(11600), oGoWsb = take(100);             // gvp_out
  size_t oGvWh = take(48), oGvWs = take(4200), oGvWsb = take(100), oGvWv = take(256); // gvp_v
  struct LOff {
    size_t ff0wh, ff0ws, ff0wsb, ff0wv, ff1wh, ff1ws, ff1wsb, ff1wv;
    size_t ln0b, ln0g, ln1b, ln1g;
    size_t m0wh, m0ws, m0wsb, m0wv, m1wh, m1ws, m1wsb, m1wv, m2wh, m2ws, m2wsb, m2wv;
  } L[3];
  for (int l = 0; l < 3; ++l) {
    L[l].ff0wh = take(32*16);  L[l].ff0ws = take(400*132); L[l].ff0wsb = take(400); L[l].ff0wv = take(32*32);
    L[l].ff1wh = take(32*32);  L[l].ff1ws = take(100*432); L[l].ff1wsb = take(100); L[l].ff1wv = take(16*32);
    L[l].ln0b = take(100); L[l].ln0g = take(100); L[l].ln1b = take(100); L[l].ln1g = take(100);
    L[l].m0wh = take(33*33); L[l].m0ws = take(100*265); L[l].m0wsb = take(100); L[l].m0wv = take(16*33);
    L[l].m1wh = take(16*16); L[l].m1ws = take(100*116); L[l].m1wsb = take(100); L[l].m1wv = take(16*16);
    L[l].m2wh = take(16*16); L[l].m2ws = take(100*116); L[l].m2wsb = take(100); L[l].m2wv = take(16*16);
  }
  size_t oLeB = take(32),  oLeG = take(32);
  size_t oLoB = take(100), oLoG = take(100);
  size_t oLvB = take(26),  oLvG = take(26);

  // --- workspace layout ---
  uintptr_t cur = (uintptr_t)d_ws;
  auto alloc = [&](size_t bytes) -> void* {
    cur = (cur + 255) & ~(uintptr_t)255;
    void* p = (void*)cur; cur += bytes; return p;
  };
  float* S    = (float*)alloc((size_t)NN*100*4);
  float* V    = (float*)alloc((size_t)NN*48*4);
  float* ESb  = (float*)alloc((size_t)NE*32*4);
  float* EVb  = (float*)alloc((size_t)NE*3*4);
  float* DSb  = (float*)alloc((size_t)NN*100*4);
  float* DVb  = (float*)alloc((size_t)NN*48*4);
  float* DEG  = (float*)alloc((size_t)NN*4);
  float* IDG  = (float*)alloc((size_t)NN*4);
  float* POOL = (float*)alloc((size_t)NG*100*4);
  float* NB   = (float*)alloc((size_t)NG*4);
  _Float16* H[3][5];
  for (int l = 0; l < 3; ++l) {
    H[l][0] = (_Float16*)alloc((size_t)112*288*2);  // msg0 ws
    H[l][1] = (_Float16*)alloc((size_t)112*128*2);  // msg1 ws
    H[l][2] = (_Float16*)alloc((size_t)112*128*2);  // msg2 ws
    H[l][3] = (_Float16*)alloc((size_t)400*160*2);  // ff0 ws
    H[l][4] = (_Float16*)alloc((size_t)112*448*2);  // ff1 ws
  }

  auto cvt = [&](const float* src, _Float16* dst, int rows, int cols, int rp, int cp) {
    int tot = rp * cp;
    k_cvt16<<<(tot + 255)/256, 256, 0, stream>>>(src, dst, rows, cols, rp, cp);
  };
  for (int l = 0; l < 3; ++l) {
    cvt(P + L[l].m0ws,  H[l][0], 100, 265, 112, 288);
    cvt(P + L[l].m1ws,  H[l][1], 100, 116, 112, 128);
    cvt(P + L[l].m2ws,  H[l][2], 100, 116, 112, 128);
    cvt(P + L[l].ff0ws, H[l][3], 400, 132, 400, 160);
    cvt(P + L[l].ff1ws, H[l][4], 100, 432, 112, 448);
  }

  hipMemsetAsync(DEG, 0, (size_t)NN*4, stream);
  k_deg<<<(NE + 255)/256, 256, 0, stream>>>(edst, DEG);
  k_invdeg<<<(NN + 255)/256, 256, 0, stream>>>(DEG, IDG);

  k_node_embed<<<(NN + 255)/256, 256, 0, stream>>>(
      hVs, hVv, seq, P + oWS, P + oLvG, P + oLvB,
      P + oGvWh, P + oGvWs, P + oGvWsb, P + oGvWv, S, V);
  k_edge_embed<<<(NE + 255)/256, 256, 0, stream>>>(
      hEs, hEv, P + oLeG, P + oLeB,
      P + oGeWh, P + oGeWs, P + oGeWsb, P + oGeWv, ESb, EVb);

  for (int l = 0; l < 3; ++l) {
    hipMemsetAsync(DSb, 0, (size_t)NN*100*4, stream);
    hipMemsetAsync(DVb, 0, (size_t)NN*48*4, stream);
    MsgW mw = { P + L[l].m0wh, P + L[l].m0wsb, P + L[l].m0wv,
                P + L[l].m1wh, P + L[l].m1wsb, P + L[l].m1wv,
                P + L[l].m2wh, P + L[l].m2wsb, P + L[l].m2wv,
                H[l][0], H[l][1], H[l][2] };
    k_msg<<<NE/32, 32, 0, stream>>>(S, V, ESb, EVb, esrc, edst, IDG, DSb, DVb, mw);
    FFW fw = { P + L[l].ff0wh, P + L[l].ff0wsb, P + L[l].ff0wv,
               P + L[l].ff1wh, P + L[l].ff1wsb, P + L[l].ff1wv,
               H[l][3], H[l][4],
               P + L[l].ln0g, P + L[l].ln0b, P + L[l].ln1g, P + L[l].ln1b };
    k_node_update<<<NN/16, 32, 0, stream>>>(S, V, DSb, DVb, fw);
  }

  hipMemsetAsync(POOL, 0, (size_t)NG*100*4, stream);
  hipMemsetAsync(NB, 0, (size_t)NG*4, stream);
  k_out_pool<<<(NN + 255)/256, 256, 0, stream>>>(
      S, V, batch, P + oLoG, P + oLoB,
      P + oGoWh, P + oGoWs, P + oGoWsb, POOL, NB);
  k_dense<<<1, 64, 0, stream>>>(POOL, NB, P + oDw1, P + oDb1, P + oDw2, P + oDb2, out);
}
